// DbrxAttention_53352083751569
// MI455X (gfx1250) — matrix operand order repaired and checked by reading, not compiled, not run on hardware
//
#include <hip/hip_runtime.h>

typedef __bf16 bf16;
typedef __attribute__((ext_vector_type(16))) __bf16 v16bf;
typedef __attribute__((ext_vector_type(8)))  __bf16 v8bf;
typedef __attribute__((ext_vector_type(4)))  __bf16 v4bf;
typedef __attribute__((ext_vector_type(8)))  float  v8f;

#define D_MODEL 6144
#define SEQ     2048
#define NH      48
#define NKV     8
#define HD      128
#define QSZ     (NH*HD)            // 6144
#define KVSZ    (NKV*HD)           // 1024
#define QKV_N   (QSZ + 2*KVSZ)     // 8192
#define ATT_SCALE 0.08838834764831845f   // 128^-0.5

static __device__ __forceinline__ v16bf cat8(v8bf lo, v8bf hi) {
  return __builtin_shufflevector(lo, hi, 0,1,2,3,4,5,6,7,8,9,10,11,12,13,14,15);
}
static __device__ __forceinline__ v8f v8f_zero() {
  v8f z = {0.f,0.f,0.f,0.f,0.f,0.f,0.f,0.f};
  return z;
}
static __device__ __forceinline__ v8f wmma_bf16(v16bf a, v16bf b, v8f c) {
  // D = A(16x32 bf16) * B(32x16 bf16) + C(16x16 f32)
  return __builtin_amdgcn_wmma_f32_16x16x32_bf16(false, a, false, b, (short)0, c, false, false);
}

// gfx1250 async global->LDS copy, 16 bytes per lane (ASYNCcnt-tracked).
// LDS address = low 32 bits of the generic shared pointer (DS-relative per
// the flat-aperture mapping).
static __device__ __forceinline__ void async_copy16(const bf16* lds_dst, const bf16* gsrc) {
  unsigned int laddr = (unsigned int)(unsigned long long)lds_dst;
  unsigned long long gaddr = (unsigned long long)gsrc;
  asm volatile("global_load_async_to_lds_b128 %0, %1, off"
               :: "v"(laddr), "v"(gaddr) : "memory");
}
static __device__ __forceinline__ void wait_async4() {
  asm volatile("s_wait_asynccnt 0x4" ::: "memory");
}
static __device__ __forceinline__ void wait_async0() {
  asm volatile("s_wait_asynccnt 0x0" ::: "memory");
}

// ---------------------------------------------------------------------------
// Tiled WMMA GEMM:  C[M,N] = A[M,K] * B[N,K]^T   (weights row-major [N][K])
// Block: 256 threads (8 waves), tile 128(M) x 64(N), K-step 32.
// A source is fp32 (converted to bf16 in-register) or bf16.
// ---------------------------------------------------------------------------
template<bool A_BF16, bool CLIP>
__global__ __launch_bounds__(256) void gemm_wmma(const void* __restrict__ Ap,
                                                 const float* __restrict__ Bw,
                                                 float* __restrict__ Cp,
                                                 int M, int N, int K) {
  constexpr int LDT = 40;                  // padded row stride (bf16): conflict-free ds_read_b128
  __shared__ bf16 sA[128*LDT];
  __shared__ bf16 sB[64*LDT];

  const int tid  = threadIdx.x;
  const int lane = tid & 31;
  const int wv   = tid >> 5;
  const int wm   = wv & 3;                 // 4 waves along M (32 rows each)
  const int wn   = wv >> 2;                // 2 waves along N (32 cols each)
  const int m0   = blockIdx.y * 128;
  const int n0   = blockIdx.x * 64;
  const int ln   = lane & 15;
  const int klo  = (lane & 16) ? 8 : 0;    // A-fragment K base (ISA 7.12.2)
  const int kb   = (lane & 16) ? 16 : 0;   // unused: B fragment also uses klo
  const int rb   = (lane & 16) ? 8 : 0;    // C-fragment row base

  const float* Af = (const float*)Ap;
  const bf16*  Ab = (const bf16*)Ap;

  v8f acc[2][2];
  #pragma unroll
  for (int i = 0; i < 2; i++)
    #pragma unroll
    for (int j = 0; j < 2; j++) acc[i][j] = v8f_zero();

  for (int k0 = 0; k0 < K; k0 += 32) {
    // ---- global -> registers ----
    float4 aR[4]; uint4 aRb[2]; float4 bR[2];
    if constexpr (!A_BF16) {
      #pragma unroll
      for (int i = 0; i < 4; i++) {
        int f = tid + i*256; int r = f >> 3; int c = (f & 7) * 4;
        aR[i] = *(const float4*)(Af + (size_t)(m0 + r)*K + k0 + c);
        if (k0 + 32 < K) __builtin_prefetch(Af + (size_t)(m0 + r)*K + k0 + 32 + c, 0, 0);
      }
    } else {
      #pragma unroll
      for (int i = 0; i < 2; i++) {
        int f = tid + i*256; int r = f >> 2; int c = (f & 3) * 8;
        aRb[i] = *(const uint4*)(Ab + (size_t)(m0 + r)*K + k0 + c);
      }
    }
    #pragma unroll
    for (int i = 0; i < 2; i++) {
      int f = tid + i*256; int r = f >> 3; int c = (f & 7) * 4;
      bR[i] = *(const float4*)(Bw + (size_t)(n0 + r)*K + k0 + c);
      if (k0 + 32 < K) __builtin_prefetch(Bw + (size_t)(n0 + r)*K + k0 + 32 + c, 0, 0);
    }
    __syncthreads();                       // previous tile fully consumed
    // ---- registers -> LDS (bf16) ----
    if constexpr (!A_BF16) {
      #pragma unroll
      for (int i = 0; i < 4; i++) {
        int f = tid + i*256; int r = f >> 3; int c = (f & 7) * 4;
        v4bf v; v[0]=(bf16)aR[i].x; v[1]=(bf16)aR[i].y; v[2]=(bf16)aR[i].z; v[3]=(bf16)aR[i].w;
        *(v4bf*)&sA[r*LDT + c] = v;
      }
    } else {
      #pragma unroll
      for (int i = 0; i < 2; i++) {
        int f = tid + i*256; int r = f >> 2; int c = (f & 3) * 8;
        *(uint4*)&sA[r*LDT + c] = aRb[i];
      }
    }
    #pragma unroll
    for (int i = 0; i < 2; i++) {
      int f = tid + i*256; int r = f >> 3; int c = (f & 7) * 4;
      v4bf v; v[0]=(bf16)bR[i].x; v[1]=(bf16)bR[i].y; v[2]=(bf16)bR[i].z; v[3]=(bf16)bR[i].w;
      *(v4bf*)&sB[r*LDT + c] = v;
    }
    __syncthreads();
    // ---- fragments + WMMA ----
    v16bf af[2], bfr[2];
    #pragma unroll
    for (int mt = 0; mt < 2; mt++) {
      int r = wm*32 + mt*16 + ln;
      af[mt] = cat8(*(const v8bf*)&sA[r*LDT + klo], *(const v8bf*)&sA[r*LDT + klo + 16]);
    }
    #pragma unroll
    for (int nt = 0; nt < 2; nt++) {
      int r = wn*32 + nt*16 + ln;
      bfr[nt] = cat8(*(const v8bf*)&sB[r*LDT + klo], *(const v8bf*)&sB[r*LDT + klo + 16]);
    }
    #pragma unroll
    for (int mt = 0; mt < 2; mt++)
      #pragma unroll
      for (int nt = 0; nt < 2; nt++)
        acc[mt][nt] = wmma_bf16(af[mt], bfr[nt], acc[mt][nt]);
  }
  // ---- epilogue ----
  #pragma unroll
  for (int mt = 0; mt < 2; mt++)
    #pragma unroll
    for (int nt = 0; nt < 2; nt++)
      #pragma unroll
      for (int r = 0; r < 8; r++) {
        float v = acc[mt][nt][r];
        if constexpr (CLIP) v = fminf(fmaxf(v, -8.0f), 8.0f);
        int gm = m0 + wm*32 + mt*16 + rb + r;
        int gn = n0 + wn*32 + nt*16 + ln;
        Cp[(size_t)gm*N + gn] = v;
      }
}

// ---------------------------------------------------------------------------
// Clipped QKV (fp32) -> RoPE'd Q/K (bf16 [h][s][d]) + V (bf16 [h][d][s]).
// ---------------------------------------------------------------------------
__global__ __launch_bounds__(256) void rope_pack(const int* __restrict__ pos_ids,
                                                 const float* __restrict__ qkv,
                                                 bf16* __restrict__ qw,
                                                 bf16* __restrict__ kw,
                                                 bf16* __restrict__ vw) {
  const long idx = (long)blockIdx.x * 256 + threadIdx.x;
  const long ROPE_N = (long)(NH + NKV) * SEQ * (HD/2);     // 56*2048*64
  if (idx < ROPE_N) {
    int  j    = (int)(idx & 63);
    long t    = idx >> 6;
    int  s    = (int)(t & (SEQ-1));
    int  slot = (int)(t >> 11);
    float pos = (float)pos_ids[s];
    float inv = __powf(500000.0f, -(float)j * (1.0f/64.0f));
    float c, sn;
    __sincosf(pos * inv, &sn, &c);
    const float* row = qkv + (size_t)s * QKV_N;
    int base = (slot < NH) ? slot*HD : QSZ + (slot - NH)*HD;
    float x1 = row[base + j], x2 = row[base + j + 64];
    bf16* dst = (slot < NH) ? qw + ((size_t)slot*SEQ + s)*HD
                            : kw + ((size_t)(slot - NH)*SEQ + s)*HD;
    dst[j]      = (bf16)(x1*c - x2*sn);
    dst[j + 64] = (bf16)(x2*c + x1*sn);
  } else {
    long i2 = idx - ROPE_N;                 // [0, 8*128*2048)
    int  s  = (int)(i2 & (SEQ-1));
    long t  = i2 >> 11;
    int  d  = (int)(t & (HD-1));
    int  h  = (int)(t >> 7);
    float x = qkv[(size_t)s*QKV_N + QSZ + KVSZ + h*HD + d];
    vw[((size_t)h*HD + d)*SEQ + s] = (bf16)x;   // transposed: [h][d][s]
  }
}

// ---------------------------------------------------------------------------
// Causal flash attention. Block = 8 waves = 128 consecutive queries of ONE
// head; K/V 32-key tiles are staged cooperatively into LDS with gfx1250
// async global->LDS copies (ASYNCcnt), double-buffered so the next tile's
// DMA overlaps the current tile's 16 WMMAs. Waves past their causal diagonal
// skip compute wave-uniformly (EXEC stays all-ones for WMMA) but keep
// participating in staging + barriers.
// ---------------------------------------------------------------------------
__global__ __launch_bounds__(256) void flash_attn(const bf16* __restrict__ qw,
                                                  const bf16* __restrict__ kw,
                                                  const bf16* __restrict__ vw,
                                                  bf16* __restrict__ ow) {
  constexpr int LDK = 136;                  // K tile row stride (bf16), conflict-free
  constexpr int LDV = 40;                   // V/P tile row stride (bf16), conflict-free
  __shared__ bf16 sK[2][32*LDK];            // 2 x 8704 B
  __shared__ bf16 sV[2][128*LDV];           // 2 x 10240 B
  __shared__ bf16 lP[8][16*40];             // per-wave P transpose scratch

  const int tid  = threadIdx.x;
  const int lane = tid & 31;
  const int wv   = tid >> 5;
  const int h    = blockIdx.x >> 4;         // 48 heads x 16 query blocks
  const int qb   = blockIdx.x & 15;
  const int hkv  = h / (NH / NKV);          // GQA: 6 heads share one KV head
  const int q0b  = qb * 128;
  const int q0   = q0b + wv * 16;
  const int ln   = lane & 15;
  const int klo  = (lane & 16) ? 8 : 0;
  const int kb   = (lane & 16) ? 16 : 0;
  const int rb   = (lane & 16) ? 8 : 0;

  const bf16* kbp = kw + (size_t)hkv * SEQ * HD;   // [key][d]
  const bf16* vbp = vw + (size_t)hkv * HD * SEQ;   // [d][key]

  // Q fragments (persist in registers)
  v16bf qf[4];
  {
    const bf16* qp = qw + ((size_t)h*SEQ + q0 + ln)*HD;
    #pragma unroll
    for (int kc = 0; kc < 4; kc++)
      qf[kc] = cat8(*(const v8bf*)(qp + kc*32 + klo), *(const v8bf*)(qp + kc*32 + klo + 16));
  }

  float rmax[8], rsum[8];
  v8f o[8];
  #pragma unroll
  for (int r = 0; r < 8; r++) { rmax[r] = -1e30f; rsum[r] = 0.f; }
  #pragma unroll
  for (int dt = 0; dt < 8; dt++) o[dt] = v8f_zero();

  // async staging of one 32-key K tile + V tile (4 x b128 per thread)
  auto stage = [&](int buf, int k0) {
    #pragma unroll
    for (int i = 0; i < 2; i++) {           // K: 32 rows x 128 d = 512 chunks
      int f = tid + i*256;
      int r = f >> 4, c = (f & 15) * 8;
      async_copy16(&sK[buf][r*LDK + c], kbp + (size_t)(k0 + r)*HD + c);
    }
    #pragma unroll
    for (int i = 0; i < 2; i++) {           // V: 128 d-rows x 32 keys = 512 chunks
      int f = tid + i*256;
      int r = f >> 2, c = (f & 3) * 8;
      async_copy16(&sV[buf][r*LDV + c], vbp + (size_t)r*SEQ + k0 + c);
    }
  };

  const int ktmax_blk = (q0b + 127) >> 5;   // block's causal limit (uniform)
  const int ktmax_me  = (q0 + 15) >> 5;     // this wave's causal limit
  int cur = 0;
  stage(0, 0);
  for (int kt = 0; kt <= ktmax_blk; kt++) {
    if (kt < ktmax_blk) { stage(cur ^ 1, (kt + 1) * 32); wait_async4(); }
    else                { wait_async0(); }
    __syncthreads();                        // stage-kt data visible to all waves

    if (kt <= ktmax_me) {
      const int k0 = kt * 32;
      // ---- S = Q * K^T (2 key-column tiles, K-dim 128 = 4 WMMA steps each)
      v8f s0 = v8f_zero(), s1 = v8f_zero();
      #pragma unroll
      for (int kc = 0; kc < 4; kc++) {
        const bf16* kr0 = &sK[cur][ln*LDK + kc*32 + klo];
        const bf16* kr1 = &sK[cur][(16 + ln)*LDK + kc*32 + klo];
        v16bf kf0 = cat8(*(const v8bf*)kr0, *(const v8bf*)(kr0 + 16));
        v16bf kf1 = cat8(*(const v8bf*)kr1, *(const v8bf*)(kr1 + 16));
        s0 = wmma_bf16(qf[kc], kf0, s0);
        s1 = wmma_bf16(qf[kc], kf1, s1);
      }
      // ---- online softmax (row stats via half-wave xor reductions)
      float p0a[8], p1a[8];
      const int kg0 = k0 + ln, kg1 = kg0 + 16;
      #pragma unroll
      for (int r = 0; r < 8; r++) {
        int qg = q0 + rb + r;
        float a = (kg0 <= qg) ? s0[r]*ATT_SCALE : -1e30f;
        float b = (kg1 <= qg) ? s1[r]*ATT_SCALE : -1e30f;
        float t = fmaxf(a, b);
        #pragma unroll
        for (int off = 1; off < 16; off <<= 1) t = fmaxf(t, __shfl_xor(t, off, 32));
        float nm    = fmaxf(rmax[r], t);
        float alpha = __expf(rmax[r] - nm);
        rmax[r] = nm;
        float p0 = __expf(a - nm), p1 = __expf(b - nm);
        float ps = p0 + p1;
        #pragma unroll
        for (int off = 1; off < 16; off <<= 1) ps += __shfl_xor(ps, off, 32);
        rsum[r] = rsum[r]*alpha + ps;
        #pragma unroll
        for (int dt = 0; dt < 8; dt++) o[dt][r] *= alpha;
        p0a[r] = p0; p1a[r] = p1;
      }
      // ---- C-layout P -> LDS -> A-layout fragment (DS in-order per wave)
      bf16* P = &lP[wv][0];
      #pragma unroll
      for (int r = 0; r < 8; r++) {
        P[(rb + r)*40 + ln]      = (bf16)p0a[r];
        P[(rb + r)*40 + 16 + ln] = (bf16)p1a[r];
      }
      asm volatile("s_wait_dscnt 0" ::: "memory");
      v16bf pf = cat8(*(const v8bf*)&P[ln*40 + klo], *(const v8bf*)&P[ln*40 + klo + 16]);
      // ---- O += P * V
      #pragma unroll
      for (int dt = 0; dt < 8; dt++) {
        const bf16* vr = &sV[cur][(dt*16 + ln)*LDV + klo];
        v16bf vf = cat8(*(const v8bf*)vr, *(const v8bf*)(vr + 16));
        o[dt] = wmma_bf16(pf, vf, o[dt]);
      }
    }
    __syncthreads();                        // everyone done with buffer `cur`
    cur ^= 1;
  }
  // ---- normalize + store attn output as bf16 [s][h*128+d]
  #pragma unroll
  for (int dt = 0; dt < 8; dt++)
    #pragma unroll
    for (int r = 0; r < 8; r++) {
      float v = o[dt][r] / rsum[r];
      int srow = q0 + rb + r;
      ow[(size_t)srow*D_MODEL + h*HD + dt*16 + ln] = (bf16)v;
    }
}

// ---------------------------------------------------------------------------
extern "C" void kernel_launch(void* const* d_in, const int* in_sizes, int n_in,
                              void* d_out, int out_size, void* d_ws, size_t ws_size,
                              hipStream_t stream) {
  const int*   pos    = (const int*)  d_in[0];
  const float* hidden = (const float*)d_in[1];   // [2048][6144]
  const float* wqkv   = (const float*)d_in[2];   // [8192][6144]
  const float* wout   = (const float*)d_in[3];   // [6144][6144]

  // workspace layout (~126 MB total)
  char* ws = (char*)d_ws;
  float* qkv  = (float*)ws;                                   // 2048*8192*4 = 64 MB
  size_t off  = (size_t)SEQ * QKV_N * sizeof(float);
  bf16* q_ws  = (bf16*)(ws + off); off += (size_t)NH  * SEQ * HD * sizeof(bf16);  // 24 MB
  bf16* k_ws  = (bf16*)(ws + off); off += (size_t)NKV * SEQ * HD * sizeof(bf16);  //  4 MB
  bf16* v_ws  = (bf16*)(ws + off); off += (size_t)NKV * SEQ * HD * sizeof(bf16);  //  4 MB
  bf16* a_ws  = (bf16*)(ws + off);                                                // 24 MB

  // 1) qkv = clip(hidden @ Wqkv^T, +-8)
  gemm_wmma<false, true><<<dim3(QKV_N/64, SEQ/128), 256, 0, stream>>>(
      hidden, wqkv, qkv, SEQ, QKV_N, D_MODEL);
  // 2) RoPE + bf16 pack (+ V transpose)
  {
    long total = (long)(NH + NKV)*SEQ*(HD/2) + (long)NKV*HD*SEQ;  // 9,437,184
    rope_pack<<<(int)(total/256), 256, 0, stream>>>(pos, qkv, q_ws, k_ws, v_ws);
  }
  // 3) causal flash attention (one head / 128 queries per block)
  flash_attn<<<NH * (SEQ/128), 256, 0, stream>>>(q_ws, k_ws, v_ws, a_ws);
  // 4) out = attn @ Wout^T
  gemm_wmma<true, false><<<dim3(D_MODEL/64, SEQ/128), 256, 0, stream>>>(
      a_ws, wout, (float*)d_out, SEQ, D_MODEL, D_MODEL);
}
